// VanillaRNN_91044716741336
// MI455X (gfx1250) — compile-verified, matrix operand order checked by reading, not run on hardware
//
#include <hip/hip_runtime.h>

// ---------------------------------------------------------------------------
// CDNA5 / gfx1250 implementation of the stacked vanilla RNN + softmax head.
// bf16 WMMA (v_wmma_f32_16x16x32_bf16), fp32 accumulate.
// Recurrence keeps its weight slice LDS-resident (staged via
// global_load_async_to_lds_b128 / ASYNCcnt) across all 512 timesteps.
// ---------------------------------------------------------------------------

typedef __attribute__((ext_vector_type(16))) __bf16 v16bf;
typedef __attribute__((ext_vector_type(8)))  float  v8f;

struct P32B { uint4 lo, hi; };  // 32 bytes == v16bf

// Load one 16x32 bf16 fragment per the CDNA5 16-bit A/B layout:
// lane L (0..15): row L, K = k0 + {0..7} and k0 + 16 + {0..7}
// lane L (16..31): row L-16, K = k0+8+{0..7} and k0+24+{0..7}
// caller passes absolute row and khalf = lane>>4.
__device__ __forceinline__ v16bf load_tile16(const __bf16* base, int ld, int row,
                                             int k0, int khalf) {
  const __bf16* p = base + (long)row * ld + k0 + khalf * 8;
  P32B pk;
  pk.lo = *reinterpret_cast<const uint4*>(p);
  pk.hi = *reinterpret_cast<const uint4*>(p + 16);
  return __builtin_bit_cast(v16bf, pk);
}

__device__ __forceinline__ v8f wmma_bf16(v16bf a, v16bf b, v8f c) {
  return __builtin_amdgcn_wmma_f32_16x16x32_bf16(false, a, false, b, (short)0, c,
                                                 false, false);
}

__device__ __forceinline__ v8f vzero8() {
  v8f z = {0.f, 0.f, 0.f, 0.f, 0.f, 0.f, 0.f, 0.f};
  return z;
}

// ---------------------------------------------------------------------------
// fp32 -> bf16 elementwise cast
// ---------------------------------------------------------------------------
__global__ void rnn_cast_f32_bf16(const float* __restrict__ in,
                                  __bf16* __restrict__ out, long n) {
  long i = (long)blockIdx.x * blockDim.x + threadIdx.x;
  if (i < n) out[i] = (__bf16)in[i];
}

// ---------------------------------------------------------------------------
// fp32 [K,N] -> bf16 transposed [N, ldt(=K)] (LDS-tiled)
// ---------------------------------------------------------------------------
__global__ void rnn_transpose_cast(const float* __restrict__ W,
                                   __bf16* __restrict__ WT,
                                   int K, int N, int ldt) {
  __shared__ float tile[32][33];
  int n = blockIdx.x * 32 + threadIdx.x;
#pragma unroll
  for (int i = 0; i < 4; ++i) {
    int k = blockIdx.y * 32 + threadIdx.y + i * 8;
    if (k < K && n < N) tile[threadIdx.y + i * 8][threadIdx.x] = W[(long)k * N + n];
  }
  __syncthreads();
  int k = blockIdx.y * 32 + threadIdx.x;
#pragma unroll
  for (int i = 0; i < 4; ++i) {
    int nn = blockIdx.x * 32 + threadIdx.y + i * 8;
    if (nn < N && k < K)
      WT[(long)nn * ldt + k] = (__bf16)tile[threadIdx.x][threadIdx.y + i * 8];
  }
}

__global__ void rnn_zero_u32(unsigned* p) {
  if (threadIdx.x == 0) *p = 0u;
}

// ---------------------------------------------------------------------------
// Shared bf16 GEMM: C[M, ldc] = A[M,K] x BT[N,K]^T (+bias), fp32 out.
// Wave tile: 2 M-tiles x 4 N-tiles of 16x16 (8 accumulators), explicit
// double-buffered K-slab fragments so a full slab of global_load_b128 stays
// in flight behind the 8 WMMAs of the previous slab.
// ---------------------------------------------------------------------------
__global__ __launch_bounds__(128) void rnn_gemm_bf16_f32(
    const __bf16* __restrict__ A, int lda,
    const __bf16* __restrict__ BT, int ldb,
    float* __restrict__ C, long ldc,
    const float* __restrict__ bias,
    int Nstore, int Nalloc, int K) {
  const int lane = threadIdx.x & 31;
  const int wave = threadIdx.x >> 5;
  const int r = lane & 15, hh = lane >> 4;
  const int m0 = blockIdx.y * 32;
  const int nb = blockIdx.x * 256 + wave * 64;

  v8f acc[2][4];
#pragma unroll
  for (int i = 0; i < 2; ++i)
#pragma unroll
    for (int j = 0; j < 4; ++j) acc[i][j] = vzero8();

  int brow[4];
#pragma unroll
  for (int j = 0; j < 4; ++j) {
    int rr = nb + j * 16 + r;
    brow[j] = (rr < Nalloc) ? rr : (Nalloc - 1);
  }

  // prologue: fragments for k0 = 0
  v16bf a0 = load_tile16(A, lda, m0 + r, 0, hh);
  v16bf a1 = load_tile16(A, lda, m0 + 16 + r, 0, hh);
  v16bf b0 = load_tile16(BT, ldb, brow[0], 0, hh);
  v16bf b1 = load_tile16(BT, ldb, brow[1], 0, hh);
  v16bf b2 = load_tile16(BT, ldb, brow[2], 0, hh);
  v16bf b3 = load_tile16(BT, ldb, brow[3], 0, hh);

  for (int k0 = 0; k0 < K - 32; k0 += 32) {
    const int kn = k0 + 32;
    // issue next slab's loads first (loadcnt covers them while WMMAs run)
    v16bf na0 = load_tile16(A, lda, m0 + r, kn, hh);
    v16bf na1 = load_tile16(A, lda, m0 + 16 + r, kn, hh);
    v16bf nb0 = load_tile16(BT, ldb, brow[0], kn, hh);
    v16bf nb1 = load_tile16(BT, ldb, brow[1], kn, hh);
    v16bf nb2 = load_tile16(BT, ldb, brow[2], kn, hh);
    v16bf nb3 = load_tile16(BT, ldb, brow[3], kn, hh);
    __builtin_prefetch(BT + (long)brow[0] * ldb + kn + 32, 0, 1);

    acc[0][0] = wmma_bf16(a0, b0, acc[0][0]);
    acc[1][0] = wmma_bf16(a1, b0, acc[1][0]);
    acc[0][1] = wmma_bf16(a0, b1, acc[0][1]);
    acc[1][1] = wmma_bf16(a1, b1, acc[1][1]);
    acc[0][2] = wmma_bf16(a0, b2, acc[0][2]);
    acc[1][2] = wmma_bf16(a1, b2, acc[1][2]);
    acc[0][3] = wmma_bf16(a0, b3, acc[0][3]);
    acc[1][3] = wmma_bf16(a1, b3, acc[1][3]);

    a0 = na0; a1 = na1; b0 = nb0; b1 = nb1; b2 = nb2; b3 = nb3;
  }
  // epilogue WMMAs for the last slab
  acc[0][0] = wmma_bf16(a0, b0, acc[0][0]);
  acc[1][0] = wmma_bf16(a1, b0, acc[1][0]);
  acc[0][1] = wmma_bf16(a0, b1, acc[0][1]);
  acc[1][1] = wmma_bf16(a1, b1, acc[1][1]);
  acc[0][2] = wmma_bf16(a0, b2, acc[0][2]);
  acc[1][2] = wmma_bf16(a1, b2, acc[1][2]);
  acc[0][3] = wmma_bf16(a0, b3, acc[0][3]);
  acc[1][3] = wmma_bf16(a1, b3, acc[1][3]);

#pragma unroll
  for (int j = 0; j < 4; ++j) {
    int n = nb + j * 16 + r;
    if (n < Nstore) {
      float bj = bias ? bias[n] : 0.f;
#pragma unroll
      for (int mt = 0; mt < 2; ++mt)
#pragma unroll
        for (int rr = 0; rr < 8; ++rr) {
          long m = m0 + mt * 16 + rr + 8 * hh;  // C layout: VGPR rr -> row rr(+8)
          C[m * ldc + n] = acc[mt][j][rr] + bj;
        }
    }
  }
}

// ---------------------------------------------------------------------------
// Persistent recurrence kernel: 64 single-wave blocks, one 16-column tile
// each. The block's weight slices (16 cols of Wh1T/Wx2T/Wh2T = 96 KB) are
// staged once into LDS with global_load_async_to_lds_b128 and stay resident
// for all T steps; per-step B fragments come from LDS (ds_load_b128).
// Two grid barriers per timestep; h-state ping-pong (bf16) in workspace.
// ---------------------------------------------------------------------------
#define RNN_BLOCKS 64

__device__ __forceinline__ void grid_barrier(unsigned* cnt, unsigned target) {
  __syncthreads();
  if (threadIdx.x == 0) {
    __threadfence();
    atomicAdd(cnt, 1u);
    while (atomicAdd(cnt, 0u) < target) __builtin_amdgcn_s_sleep(2);
  }
  __syncthreads();
  __threadfence();
}

__global__ __launch_bounds__(32) void rnn_steps(
    const __bf16* __restrict__ Wh1T, const __bf16* __restrict__ Wx2T,
    const __bf16* __restrict__ Wh2T, const float* __restrict__ Xp,
    const float* __restrict__ b1, const float* __restrict__ b2,
    __bf16* __restrict__ hstate, __bf16* __restrict__ H2,
    unsigned* __restrict__ cnt, int T) {
  extern __shared__ __bf16 smem[];  // 3 x [16][1024] bf16 = 96 KB, base off 0
  const int lane = threadIdx.x & 31;
  const int n0 = blockIdx.x * 16;  // this block's 16-column tile
  const int r = lane & 15, hh = lane >> 4;
  const int ncol = n0 + r;

  __bf16* h1buf = hstate;                  // 2 x [16,1024]
  __bf16* h2buf = hstate + 2 * 16 * 1024;  // 2 x [16,1024]

  // Stage this block's weight slices into LDS (async-to-LDS, ASYNCcnt).
  // No static __shared__ in this kernel -> dynamic LDS base offset is 0,
  // so the raw LDS byte offset in VDST matches the C++-side smem reads.
  {
    const __bf16* srcs[3] = {Wh1T + (long)n0 * 1024, Wx2T + (long)n0 * 1024,
                             Wh2T + (long)n0 * 1024};
    for (int mtx = 0; mtx < 3; ++mtx) {
      const char* g = (const char*)srcs[mtx];
      unsigned ldsbase = (unsigned)mtx * 32768u;
      for (int j = 0; j < 64; ++j) {  // 64 iters x 32 lanes x 16B = 32 KB
        unsigned off = ((unsigned)j * 32u + (unsigned)lane) * 16u;
        unsigned ldsaddr = ldsbase + off;
        unsigned long long ga = (unsigned long long)(g + off);
        asm volatile("global_load_async_to_lds_b128 %0, %1, off"
                     :: "v"(ldsaddr), "v"(ga) : "memory");
      }
    }
    asm volatile("s_wait_asynccnt 0x0" ::: "memory");
  }
  const __bf16* lWh1 = smem;                // [16][1024]
  const __bf16* lWx2 = smem + 16 * 1024;
  const __bf16* lWh2 = smem + 32 * 1024;

  // zero both ping-pong state buffers (h(0) = 0, and clear 0xAA poison)
  for (int i = blockIdx.x * 32 + threadIdx.x; i < 4 * 16 * 1024;
       i += RNN_BLOCKS * 32)
    hstate[i] = (__bf16)0.f;

  unsigned phase = 0;
  grid_barrier(cnt, ++phase * RNN_BLOCKS);

  const float bias1 = b1[ncol];
  const float bias2 = b2[ncol];

  for (int t = 0; t < T; ++t) {
    const int cur = t & 1, nxt = cur ^ 1;
    const __bf16* h1c = h1buf + cur * 16384;
    __bf16* h1n = h1buf + nxt * 16384;
    const __bf16* h2c = h2buf + cur * 16384;
    __bf16* h2n = h2buf + nxt * 16384;

    // ---- layer 1: h1n = tanh(Xp[:,t] + h1c @ Wh1 + b1), tile [16 x 16] ----
    v8f c = vzero8();
    for (int k0 = 0; k0 < 1024; k0 += 32) {
      v16bf a = load_tile16(h1c, 1024, r, k0, hh);       // global (L2-hot)
      v16bf b = load_tile16(lWh1, 1024, r, k0, hh);      // LDS-resident
      c = wmma_bf16(a, b, c);
    }
#pragma unroll
    for (int rr = 0; rr < 8; ++rr) {
      int m = rr + 8 * hh;  // C row this accumulator register holds
      float x = (m < 8) ? Xp[((long)m * T + t) * 1024 + ncol] : 0.f;
      float v = tanhf(c[rr] + x + bias1);
      h1n[m * 1024 + ncol] = (__bf16)v;
    }
    grid_barrier(cnt, ++phase * RNN_BLOCKS);

    // ---- layer 2: h2n = tanh(h1n @ Wx2 + h2c @ Wh2 + b2) ----
    c = vzero8();
    for (int k0 = 0; k0 < 1024; k0 += 32) {
      v16bf a = load_tile16(h1n, 1024, r, k0, hh);
      v16bf b = load_tile16(lWx2, 1024, r, k0, hh);
      c = wmma_bf16(a, b, c);
    }
    for (int k0 = 0; k0 < 1024; k0 += 32) {
      v16bf a = load_tile16(h2c, 1024, r, k0, hh);
      v16bf b = load_tile16(lWh2, 1024, r, k0, hh);
      c = wmma_bf16(a, b, c);
    }
#pragma unroll
    for (int rr = 0; rr < 8; ++rr) {
      int m = rr + 8 * hh;
      float v = tanhf(c[rr] + bias2);
      h2n[m * 1024 + ncol] = (__bf16)v;
      if (m < 8) H2[((long)m * T + t) * 1024 + ncol] = (__bf16)v;
    }
    grid_barrier(cnt, ++phase * RNN_BLOCKS);
  }
}

// ---------------------------------------------------------------------------
// In-place row softmax over V elements, one block per row, online max/sum.
// ---------------------------------------------------------------------------
__global__ __launch_bounds__(256) void rnn_softmax_rows(float* __restrict__ out,
                                                        int V) {
  float* p = out + (long)blockIdx.x * V;
  const int tid = threadIdx.x;

  float m = -__builtin_inff(), s = 0.f;
  for (int i = tid; i < V; i += 256) {
    float x = p[i];
    if (x > m) {
      s = s * __expf(m - x) + 1.f;
      m = x;
    } else {
      s += __expf(x - m);
    }
  }
  __shared__ float sm[256], ss[256];
  sm[tid] = m;
  ss[tid] = s;
  __syncthreads();
  for (int off = 128; off > 0; off >>= 1) {
    if (tid < off) {
      float m2 = sm[tid + off], s2 = ss[tid + off];
      float M = fmaxf(sm[tid], m2);
      ss[tid] = ss[tid] * __expf(sm[tid] - M) + s2 * __expf(m2 - M);
      sm[tid] = M;
    }
    __syncthreads();
  }
  float M = sm[0];
  float inv = 1.f / ss[0];
  for (int i = tid; i < V; i += 256) p[i] = __expf(p[i] - M) * inv;
}

// ---------------------------------------------------------------------------
// Host orchestration
// ---------------------------------------------------------------------------
extern "C" void kernel_launch(void* const* d_in, const int* in_sizes, int n_in,
                              void* d_out, int out_size, void* d_ws,
                              size_t ws_size, hipStream_t stream) {
  (void)in_sizes; (void)n_in; (void)out_size; (void)ws_size;
  constexpr int T = 512, D = 512, U = 1024;
  constexpr int V = 50257;
  constexpr int M = 8 * T;                  // 4096 rows (b*T + t)
  constexpr int Vp = ((V + 15) / 16) * 16;  // 50272

  const float* inputs = (const float*)d_in[0];
  const float* W_x1   = (const float*)d_in[1];
  const float* W_h1   = (const float*)d_in[2];
  const float* b1     = (const float*)d_in[3];
  const float* W_x2   = (const float*)d_in[4];
  const float* W_h2   = (const float*)d_in[5];
  const float* b2     = (const float*)d_in[6];
  const float* W_out  = (const float*)d_in[7];
  const float* b_out  = (const float*)d_in[8];
  float* out = (float*)d_out;

  char* base = (char*)d_ws;
  size_t off = 0;
  auto take = [&](size_t bytes) -> void* {
    size_t a = (off + 255) & ~(size_t)255;
    off = a + bytes;
    return (void*)(base + a);
  };

  __bf16* Xbf    = (__bf16*)take((size_t)M * D * 2);
  __bf16* Wx1T   = (__bf16*)take((size_t)U * D * 2);
  __bf16* Wh1T   = (__bf16*)take((size_t)U * U * 2);
  __bf16* Wx2T   = (__bf16*)take((size_t)U * U * 2);
  __bf16* Wh2T   = (__bf16*)take((size_t)U * U * 2);
  __bf16* WoutT  = (__bf16*)take((size_t)Vp * U * 2);  // ~103 MB, L2-resident
  float*  Xp     = (float*)take((size_t)M * U * 4);
  __bf16* H2     = (__bf16*)take((size_t)M * U * 2);
  __bf16* hstate = (__bf16*)take((size_t)4 * 16 * 1024 * 2);
  unsigned* cnt  = (unsigned*)take(256);

  // 1. cast inputs to bf16
  {
    long n = (long)M * D;
    rnn_cast_f32_bf16<<<dim3((unsigned)((n + 255) / 256)), dim3(256), 0,
                        stream>>>(inputs, Xbf, n);
  }
  // 2. transpose+cast weights: [K,N] fp32 -> [N,K] bf16
  dim3 tb(32, 8);
  rnn_transpose_cast<<<dim3((U + 31) / 32, (D + 31) / 32), tb, 0, stream>>>(
      W_x1, Wx1T, D, U, D);
  rnn_transpose_cast<<<dim3((U + 31) / 32, (U + 31) / 32), tb, 0, stream>>>(
      W_h1, Wh1T, U, U, U);
  rnn_transpose_cast<<<dim3((U + 31) / 32, (U + 31) / 32), tb, 0, stream>>>(
      W_x2, Wx2T, U, U, U);
  rnn_transpose_cast<<<dim3((U + 31) / 32, (U + 31) / 32), tb, 0, stream>>>(
      W_h2, Wh2T, U, U, U);
  rnn_transpose_cast<<<dim3((V + 31) / 32, (U + 31) / 32), tb, 0, stream>>>(
      W_out, WoutT, U, V, U);
  // 3. reset grid barrier counter (must not rely on ws state across calls)
  rnn_zero_u32<<<1, 32, 0, stream>>>(cnt);
  // 4. input projection for all timesteps: Xp = Xbf @ Wx1  (M x U)
  rnn_gemm_bf16_f32<<<dim3(U / 256, M / 32), 128, 0, stream>>>(
      Xbf, D, Wx1T, D, Xp, U, nullptr, U, U, D);
  // 5. sequential recurrence (persistent, 64 single-wave blocks, 96 KB LDS)
  rnn_steps<<<dim3(RNN_BLOCKS), 32, 96 * 1024, stream>>>(
      Wh1T, Wx2T, Wh2T, Xp, b1, b2, hstate, H2, cnt, T);
  // 6. head GEMM: logits = H2 @ WoutT^T + b_out  ->  d_out (row = b*T + t)
  rnn_gemm_bf16_f32<<<dim3((V + 255) / 256, M / 32), 128, 0, stream>>>(
      H2, U, WoutT, U, out, V, b_out, V, Vp, U);
  // 7. in-place softmax per row
  rnn_softmax_rows<<<dim3(M), 256, 0, stream>>>(out, V);
}
